// SpinTransformerModule_62869731279058
// MI455X (gfx1250) — matrix operand order repaired and checked by reading, not compiled, not run on hardware
//
#include <hip/hip_runtime.h>
#include <hip/hip_bf16.h>
#include <math.h>

// ---------------- constants ----------------
#define B_      2
#define N_      2048
#define DIM_    1024
#define HEADS_  16
#define DH_     64
#define SCALE_F       22.60530911040405f     // sqrt(511) = sqrt(DIM/2-1)
#define SCALE_HEAD_F  5.5677643628300215f    // sqrt(31)  = sqrt(DH/2-1)
#define INV_SCALE2    (1.0f/511.0f)

typedef __attribute__((ext_vector_type(16))) __bf16 v16bf;
typedef __attribute__((ext_vector_type(8)))  __bf16 v8bf;
typedef __attribute__((ext_vector_type(8)))  float  v8f;
typedef __attribute__((ext_vector_type(4)))  unsigned int u32x4;
typedef __attribute__((ext_vector_type(8)))  int   i32x8;
typedef __attribute__((ext_vector_type(4)))  int   i32x4;

static __device__ __forceinline__ v16bf cat8(v8bf lo, v8bf hi) {
    return __builtin_shufflevector(lo, hi, 0,1,2,3,4,5,6,7,8,9,10,11,12,13,14,15);
}
static __device__ __forceinline__ v8f wmma_bf(v16bf a, v16bf b, v8f c) {
    return __builtin_amdgcn_wmma_f32_16x16x32_bf16(false, a, false, b, (short)0, c, false, false);
}
// sum across each 16-lane row group via DPP row_ror rotate-and-add (pure VALU, no LDS)
static __device__ __forceinline__ float half16_sum(float x) {
    float t;
    t = __builtin_bit_cast(float, __builtin_amdgcn_update_dpp(0, __builtin_bit_cast(int, x), 0x121, 0xf, 0xf, true)); x += t; // row_ror:1
    t = __builtin_bit_cast(float, __builtin_amdgcn_update_dpp(0, __builtin_bit_cast(int, x), 0x122, 0xf, 0xf, true)); x += t; // row_ror:2
    t = __builtin_bit_cast(float, __builtin_amdgcn_update_dpp(0, __builtin_bit_cast(int, x), 0x124, 0xf, 0xf, true)); x += t; // row_ror:4
    t = __builtin_bit_cast(float, __builtin_amdgcn_update_dpp(0, __builtin_bit_cast(int, x), 0x128, 0xf, 0xf, true)); x += t; // row_ror:8
    return x;
}

// ---------------- Tensor Data Mover: 2D tile global -> LDS ----------------
// D# per CDNA5 ISA ch.8: group0 {count,lds_addr,global_addr,type=2},
// group1 {data_size=2B, pad 4 DWORDs every 16 DWORDs (LDS row stride 40 halves),
//         tensor dims, tile 32x(rows), dim0 stride}.
static __device__ __forceinline__ void tdm_load_tile(unsigned int lds_off, const void* gaddr,
                                                     unsigned int rows, unsigned int tensor_w,
                                                     unsigned int stride_elems) {
    unsigned long long ga = (unsigned long long)(uintptr_t)gaddr;
    u32x4 g0;
    g0[0] = 1u;                                                   // count=1 valid descriptor
    g0[1] = lds_off;                                              // LDS byte address
    g0[2] = (unsigned int)(ga & 0xFFFFFFFFu);                     // global_addr[31:0]
    g0[3] = (unsigned int)((ga >> 32) & 0x01FFFFFFu) | (2u << 30);// global_addr[56:32], type=2
    i32x8 g1;
    g1[0] = (int)((1u << 16) | (1u << 20) | (3u << 22) | (3u << 25)); // data_size=2B, pad_en, intv=16dw, amt=4dw
    g1[1] = (int)((tensor_w & 0xFFFFu) << 16);                        // tensor_dim0[15:0]
    g1[2] = (int)(((tensor_w >> 16) & 0xFFFFu) | ((rows & 0xFFFFu) << 16)); // dim0 hi | tensor_dim1 lo
    g1[3] = (int)(((rows >> 16) & 0xFFFFu) | (32u << 16));            // dim1 hi | tile_dim0=32
    g1[4] = (int)rows;                                                // tile_dim1, tile_dim2=0
    g1[5] = (int)stride_elems;                                        // tensor_dim0_stride[31:0]
    g1[6] = 0;
    g1[7] = 0;
    i32x4 z4; z4[0] = z4[1] = z4[2] = z4[3] = 0;
#if defined(__clang_major__) && (__clang_major__ >= 23)
    i32x8 z8; for (int i = 0; i < 8; ++i) z8[i] = 0;
    __builtin_amdgcn_tensor_load_to_lds(g0, g1, z4, z4, z8, 0);
#else
    __builtin_amdgcn_tensor_load_to_lds(g0, g1, z4, z4, 0);
#endif
}

// ---------------- small elementwise kernels ----------------
__global__ void k_f32_to_bf16(const float* __restrict__ in, __bf16* __restrict__ out, int n) {
    int i = blockIdx.x * blockDim.x + threadIdx.x;
    int stride = gridDim.x * blockDim.x;
    for (; i < n; i += stride) out[i] = (__bf16)in[i];
}

// xs = SCALE * normalize(x, dim=-1); store f32 and bf16
__global__ __launch_bounds__(256) void k_norm_x(const float* __restrict__ x,
                                                float* __restrict__ xs_f,
                                                __bf16* __restrict__ xs_bf) {
    const int row = blockIdx.x;            // 0..4095
    const int t   = threadIdx.x;           // 256 threads, 4 elems each
    const float* xr = x + (size_t)row * DIM_;
    float v[4]; float ss = 0.f;
#pragma unroll
    for (int i = 0; i < 4; ++i) { v[i] = xr[t*4 + i]; ss += v[i]*v[i]; }
    ss = half16_sum(ss);
    ss += __shfl_xor(ss, 16, 32);
    __shared__ float wsum[8];
    if ((t & 31) == 0) wsum[t >> 5] = ss;
    __syncthreads();
    float tot = 0.f;
#pragma unroll
    for (int i = 0; i < 8; ++i) tot += wsum[i];
    const float s = SCALE_F / fmaxf(sqrtf(tot), 1e-12f);
#pragma unroll
    for (int i = 0; i < 4; ++i) {
        float o = v[i] * s;
        xs_f [(size_t)row * DIM_ + t*4 + i] = o;
        xs_bf[(size_t)row * DIM_ + t*4 + i] = (__bf16)o;
    }
}

// per-head normalize q,k from qk[4096,2048]; write bf16 [b,h,n,64]; fold sqrt(31) into q
__global__ __launch_bounds__(256) void k_qknorm(const float* __restrict__ qk,
                                                __bf16* __restrict__ qn,
                                                __bf16* __restrict__ kn) {
    const int w  = threadIdx.x >> 5;
    const int l  = threadIdx.x & 31;
    const int vec = blockIdx.x * 8 + w;    // 4096 rows * 16 heads * 2 = 131072
    const int row = vec >> 5;
    const int rem = vec & 31;
    const int h   = rem >> 1;
    const int isK = rem & 1;
    const int b = row >> 11, n = row & (N_-1);
    const float* src = qk + (size_t)row * (2*DIM_) + isK*DIM_ + h*DH_;
    float a0 = src[l*2 + 0], a1 = src[l*2 + 1];
    float ss = a0*a0 + a1*a1;
    ss = half16_sum(ss);
    ss += __shfl_xor(ss, 16, 32);
    const float scl = (isK ? 1.0f : SCALE_HEAD_F) / fmaxf(sqrtf(ss), 1e-12f);
    __bf16* dst = (isK ? kn : qn) + (((size_t)(b*HEADS_ + h)) * N_ + n) * DH_;
    dst[l*2 + 0] = (__bf16)(a0 * scl);
    dst[l*2 + 1] = (__bf16)(a1 * scl);
}

// ---------------- dense GEMM C = A[M,K] * W[Nout,K]^T, TDM-staged, double-buffered ----------------
enum { MODE_QK = 0, MODE_GELU = 1, MODE_FF = 2 };
#define LDS_STRIDE 40   // 32 halves + 4-DWORD TDM pad per row: conflict-free b128 frag reads

template <int MODE>
__global__ __launch_bounds__(256) void k_gemm(const __bf16* __restrict__ A,
                                              const __bf16* __restrict__ Wm,
                                              void* __restrict__ Out,
                                              const float* __restrict__ aux,
                                              int Nout, int K) {
    __shared__ __align__(16) __bf16 Ash[2][128 * LDS_STRIDE];
    __shared__ __align__(16) __bf16 Bsh[2][128 * LDS_STRIDE];
    const int t  = threadIdx.x;
    const int w  = t >> 5, l = t & 31, ln = l & 15, hs = l >> 4;
    const int wr = w & 3, wc = w >> 2;               // 4x2 wave grid, wave tile 32x64
    const int bm = blockIdx.y * 128, bn = blockIdx.x * 128;

    v8f acc[2][4];
#pragma unroll
    for (int mt = 0; mt < 2; ++mt)
#pragma unroll
        for (int nt = 0; nt < 4; ++nt)
#pragma unroll
            for (int e = 0; e < 8; ++e) acc[mt][nt][e] = 0.f;

    const int nk = K >> 5;
    // prologue: DMA tile 0 into buffer 0 (one TDM op per tile, wave 0 only)
    if (w == 0) {
        tdm_load_tile((unsigned int)(uintptr_t)&Ash[0][0], A  + (size_t)bm * K, 128, (unsigned int)K, (unsigned int)K);
        tdm_load_tile((unsigned int)(uintptr_t)&Bsh[0][0], Wm + (size_t)bn * K, 128, (unsigned int)K, (unsigned int)K);
    }
    for (int ks = 0; ks < nk; ++ks) {
        const int cur = ks & 1;
        if (w == 0) {
            if (ks + 1 < nk) {   // prefetch next K-tile into the other buffer
                int kk = (ks + 1) << 5;
                tdm_load_tile((unsigned int)(uintptr_t)&Ash[cur ^ 1][0], A  + (size_t)bm * K + kk, 128, (unsigned int)K, (unsigned int)K);
                tdm_load_tile((unsigned int)(uintptr_t)&Bsh[cur ^ 1][0], Wm + (size_t)bn * K + kk, 128, (unsigned int)K, (unsigned int)K);
                __builtin_amdgcn_s_wait_tensorcnt(2);  // current buffer's 2 DMAs done
            } else {
                __builtin_amdgcn_s_wait_tensorcnt(0);
            }
        }
        __syncthreads();

        v16bf af[2], bfr[4];
#pragma unroll
        for (int mt = 0; mt < 2; ++mt) {
            int r = wr*32 + mt*16 + ln;
            const __bf16* p = &Ash[cur][r * LDS_STRIDE + hs*16];
            af[mt] = cat8(*(const v8bf*)p, *(const v8bf*)(p + 8));
        }
#pragma unroll
        for (int nt = 0; nt < 4; ++nt) {
            int cidx = wc*64 + nt*16 + ln;
            const __bf16* p = &Bsh[cur][cidx * LDS_STRIDE + hs*16];
            bfr[nt] = cat8(*(const v8bf*)p, *(const v8bf*)(p + 8));
        }
#pragma unroll
        for (int mt = 0; mt < 2; ++mt)
#pragma unroll
            for (int nt = 0; nt < 4; ++nt)
                acc[mt][nt] = wmma_bf(af[mt], bfr[nt], acc[mt][nt]);
        __syncthreads();   // all waves done with buf[cur] before it becomes a DMA target
    }

    // epilogue: C-tile lane layout: row = v + 8*hs, col = ln
#pragma unroll
    for (int mt = 0; mt < 2; ++mt)
#pragma unroll
        for (int nt = 0; nt < 4; ++nt)
#pragma unroll
            for (int v = 0; v < 8; ++v) {
                int row = bm + wr*32 + mt*16 + v + 8*hs;
                int col = bn + wc*64 + nt*16 + ln;
                float x = acc[mt][nt][v];
                if (MODE == MODE_QK) {
                    ((float*)Out)[(size_t)row * Nout + col] = x;
                } else if (MODE == MODE_GELU) {
                    float g = 0.5f * x * (1.0f + erff(x * 0.70710678118654752f));
                    ((__bf16*)Out)[(size_t)row * Nout + col] = (__bf16)g;
                } else { // MODE_FF: xf[b,h,n,d] = ff + xs
                    int b = row >> 11, n = row & (N_-1);
                    int h = col >> 6,  d = col & (DH_-1);
                    ((float*)Out)[(((size_t)(b*HEADS_ + h)) * N_ + n) * DH_ + d] =
                        x + aux[(size_t)row * DIM_ + col];
                }
            }
}

// -------- softmax row denominators: rinv[row] = 1 / sum_j exp(sim[row,j]) --------
// safe without max-subtraction: |sim| <= sqrt(31), exp <= 262, sums <= ~5.4e5.
// Per-lane partials in the loop; one DPP reduction at the end (no LDS in the hot loop).
__global__ __launch_bounds__(256) void k_rowsum(const __bf16* __restrict__ qn,
                                                const __bf16* __restrict__ kn,
                                                float* __restrict__ rinv) {
    const int w  = threadIdx.x >> 5;
    const int l  = threadIdx.x & 31, ln = l & 15, hs = l >> 4;
    const int bh = blockIdx.x >> 4;
    const int rb = blockIdx.x & 15;
    const int rowbase = rb * 128 + w * 16;
    const __bf16* qb = qn + (size_t)bh * N_ * DH_;
    const __bf16* kb = kn + (size_t)bh * N_ * DH_;

    v16bf aq[2];
    {
        int arow = rowbase + ln;
#pragma unroll
        for (int kd = 0; kd < 2; ++kd) {
            const __bf16* p = qb + (size_t)arow * DH_ + kd*32 + hs*16;
            aq[kd] = cat8(*(const v8bf*)p, *(const v8bf*)(p + 8));
        }
    }
    float rsum[8];
#pragma unroll
    for (int v = 0; v < 8; ++v) rsum[v] = 0.f;

    for (int jb = 0; jb < N_; jb += 32) {
#pragma unroll
        for (int jn = 0; jn < 2; ++jn) {
            v8f c;
#pragma unroll
            for (int e = 0; e < 8; ++e) c[e] = 0.f;
#pragma unroll
            for (int kd = 0; kd < 2; ++kd) {
                int j = jb + jn*16 + ln;
                const __bf16* p = kb + (size_t)j * DH_ + kd*32 + hs*16;
                v16bf bk = cat8(*(const v8bf*)p, *(const v8bf*)(p + 8));
                c = wmma_bf(aq[kd], bk, c);
            }
#pragma unroll
            for (int v = 0; v < 8; ++v) rsum[v] += __expf(c[v]);
        }
    }
#pragma unroll
    for (int v = 0; v < 8; ++v) {
        float s = half16_sum(rsum[v]);
        if (ln == 0)
            rinv[(size_t)bh * N_ + rowbase + v + 8*hs] = 1.0f / s;
    }
}

// ---------------- mean-field iteration step (flash-style attn apply) ----------------
// PHASE: 0 = first (m := const m0), 1 = middle, 2 = last (write f32 output)
template <int PHASE>
__global__ __launch_bounds__(256) void k_iter(const __bf16* __restrict__ qn,
                                              const __bf16* __restrict__ kn,
                                              const float*  __restrict__ xf,
                                              const float*  __restrict__ rinv,
                                              const __bf16* __restrict__ mIn,   // mT [b,h,64,N]
                                              __bf16* __restrict__ mOut,        // mT [b,h,64,N]
                                              float* __restrict__ outF) {       // [B,N,DIM]
    __shared__ __align__(16) __bf16 pbuf[8 * 16 * 32];   // wave-private p tiles
    const int  w  = threadIdx.x >> 5;
    const int  l  = threadIdx.x & 31, ln = l & 15, hs = l >> 4;
    const int  bh = blockIdx.x >> 4;
    const int  rb = blockIdx.x & 15;
    const int  rowbase = rb * 128 + w * 16;
    const __bf16* qb = qn + (size_t)bh * N_ * DH_;
    const __bf16* kb = kn + (size_t)bh * N_ * DH_;
    __bf16* pw = &pbuf[w * 512];

    v16bf aq[2];
    {
        int arow = rowbase + ln;
#pragma unroll
        for (int kd = 0; kd < 2; ++kd) {
            const __bf16* p = qb + (size_t)arow * DH_ + kd*32 + hs*16;
            aq[kd] = cat8(*(const v8bf*)p, *(const v8bf*)(p + 8));
        }
    }
    float rinv_r[8];
#pragma unroll
    for (int v = 0; v < 8; ++v)
        rinv_r[v] = rinv[(size_t)bh * N_ + rowbase + v + 8*hs];

    v8f acc[4];
#pragma unroll
    for (int dt = 0; dt < 4; ++dt)
#pragma unroll
        for (int e = 0; e < 8; ++e) acc[dt][e] = 0.f;

    v16bf mconst;
    if (PHASE == 0) {
        const float m0 = 1.0f / (1.0f + sqrtf(1.0f + (float)DH_ * INV_SCALE2));
#pragma unroll
        for (int e = 0; e < 16; ++e) mconst[e] = (__bf16)m0;
    }

    for (int jb = 0; jb < N_; jb += 32) {
        // s = qn(16 rows) x kn(32 cols)^T   (K = dh = 64); p = exp(s) (rinv applied in epilogue)
#pragma unroll
        for (int jn = 0; jn < 2; ++jn) {
            v8f c;
#pragma unroll
            for (int e = 0; e < 8; ++e) c[e] = 0.f;
#pragma unroll
            for (int kd = 0; kd < 2; ++kd) {
                int j = jb + jn*16 + ln;
                const __bf16* p = kb + (size_t)j * DH_ + kd*32 + hs*16;
                v16bf bk = cat8(*(const v8bf*)p, *(const v8bf*)(p + 8));
                c = wmma_bf(aq[kd], bk, c);
            }
#pragma unroll
            for (int v = 0; v < 8; ++v)
                pw[(v + 8*hs) * 32 + jn*16 + ln] = (__bf16)__expf(c[v]);
        }
        __builtin_amdgcn_wave_barrier();   // DS ops are in-order per wave; stop compiler reordering
        v16bf pa;
        {
            const __bf16* p = &pw[ln * 32 + hs*16];
            pa = cat8(*(const v8bf*)p, *(const v8bf*)(p + 8));
        }
        // acc[16 rows, 64 d] += p[16,32] x m[32,64]
#pragma unroll
        for (int dt = 0; dt < 4; ++dt) {
            v16bf bm;
            if (PHASE == 0) {
                bm = mconst;
            } else {
                int d = dt*16 + ln;
                const __bf16* p = mIn + ((size_t)bh * DH_ + d) * N_ + jb + hs*16;
                bm = cat8(*(const v8bf*)p, *(const v8bf*)(p + 8));
            }
            acc[dt] = wmma_bf(pa, bm, acc[dt]);
        }
        __builtin_amdgcn_wave_barrier();
    }

    // epilogue: theta = rinv*acc + xf; m = theta / (1 + sqrt(1 + |theta|^2/511))
#pragma unroll
    for (int dt = 0; dt < 4; ++dt)
#pragma unroll
        for (int v = 0; v < 8; ++v) {
            int row = rowbase + v + 8*hs;
            acc[dt][v] = acc[dt][v] * rinv_r[v]
                       + xf[((size_t)bh * N_ + row) * DH_ + dt*16 + ln];
        }
#pragma unroll
    for (int v = 0; v < 8; ++v) {
        float h2 = 0.f;
#pragma unroll
        for (int dt = 0; dt < 4; ++dt) h2 += acc[dt][v] * acc[dt][v];
        h2 = half16_sum(h2);
        const float inv = 1.0f / (1.0f + sqrtf(1.0f + h2 * INV_SCALE2));
        const int row = rowbase + v + 8*hs;
#pragma unroll
        for (int dt = 0; dt < 4; ++dt) {
            float mv = acc[dt][v] * inv;
            int d = dt*16 + ln;
            if (PHASE == 2) {
                int b = bh >> 4, h = bh & (HEADS_-1);
                outF[((size_t)b * N_ + row) * DIM_ + h*DH_ + d] = mv;
            } else {
                mOut[((size_t)bh * DH_ + d) * N_ + row] = (__bf16)mv;
            }
        }
    }
}

// ---------------- host launcher ----------------
extern "C" void kernel_launch(void* const* d_in, const int* in_sizes, int n_in,
                              void* d_out, int out_size, void* d_ws, size_t ws_size,
                              hipStream_t stream) {
    (void)in_sizes; (void)n_in; (void)out_size; (void)ws_size;
    const float* x    = (const float*)d_in[0];
    const float* w_qk = (const float*)d_in[1];
    const float* w1   = (const float*)d_in[2];
    const float* w2   = (const float*)d_in[3];

    char* ws = (char*)d_ws;
    auto alloc = [&](size_t bytes) -> char* {
        char* p = ws;
        ws += (bytes + 255) & ~(size_t)255;
        return p;
    };
    __bf16* wqk_bf = (__bf16*)alloc((size_t)2048*1024*2);
    __bf16* w1_bf  = (__bf16*)alloc((size_t)4096*1024*2);
    __bf16* w2_bf  = (__bf16*)alloc((size_t)1024*4096*2);
    __bf16* xs_bf  = (__bf16*)alloc((size_t)4096*1024*2);
    float*  xs_f   = (float*) alloc((size_t)4096*1024*4);
    char*   qk_reg = alloc((size_t)4096*2048*4);     // 32MB: qk f32, later reused as h bf16
    float*  qk_f   = (float*)qk_reg;
    __bf16* h_bf   = (__bf16*)qk_reg;
    __bf16* qn     = (__bf16*)alloc((size_t)B_*HEADS_*N_*DH_*2);
    __bf16* kn     = (__bf16*)alloc((size_t)B_*HEADS_*N_*DH_*2);
    float*  xf     = (float*) alloc((size_t)B_*HEADS_*N_*DH_*4);
    float*  rinv   = (float*) alloc((size_t)B_*HEADS_*N_*4);
    __bf16* mA     = (__bf16*)alloc((size_t)B_*HEADS_*N_*DH_*2);
    __bf16* mB     = (__bf16*)alloc((size_t)B_*HEADS_*N_*DH_*2);

    // 1) weights f32 -> bf16
    k_f32_to_bf16<<<4096, 256, 0, stream>>>(w_qk, wqk_bf, 2048*1024);
    k_f32_to_bf16<<<4096, 256, 0, stream>>>(w1,   w1_bf,  4096*1024);
    k_f32_to_bf16<<<4096, 256, 0, stream>>>(w2,   w2_bf,  1024*4096);
    // 2) xs = sqrt(511)*normalize(x)
    k_norm_x<<<4096, 256, 0, stream>>>(x, xs_f, xs_bf);
    // 3) qk = xs @ w_qk^T   [4096, 2048]
    k_gemm<MODE_QK><<<dim3(2048/128, 4096/128), 256, 0, stream>>>(xs_bf, wqk_bf, qk_f, nullptr, 2048, 1024);
    // 4) per-head normalize q,k (fold sqrt(31) into q)
    k_qknorm<<<16384, 256, 0, stream>>>(qk_f, qn, kn);
    // 5) h = gelu(xs @ w1^T)  [4096,4096] bf16  (reuses qk region)
    k_gemm<MODE_GELU><<<dim3(4096/128, 4096/128), 256, 0, stream>>>(xs_bf, w1_bf, h_bf, nullptr, 4096, 1024);
    // 6) xf = xh + h @ w2^T, stored [b,h,n,64] f32
    k_gemm<MODE_FF><<<dim3(1024/128, 4096/128), 256, 0, stream>>>(h_bf, w2_bf, xf, xs_f, 1024, 4096);
    // 7) softmax row denominators (reciprocal)
    k_rowsum<<<B_*HEADS_*(N_/128), 256, 0, stream>>>(qn, kn, rinv);
    // 8) 8 mean-field steps (ping-pong mT bf16; last writes f32 output)
    const int NBLK = B_*HEADS_*(N_/128);
    k_iter<0><<<NBLK, 256, 0, stream>>>(qn, kn, xf, rinv, mB, mA, (float*)d_out);
    for (int s = 1; s < 7; ++s) {
        const __bf16* mi = (s & 1) ? mA : mB;
        __bf16*       mo = (s & 1) ? mB : mA;
        k_iter<1><<<NBLK, 256, 0, stream>>>(qn, kn, xf, rinv, mi, mo, (float*)d_out);
    }
    k_iter<2><<<NBLK, 256, 0, stream>>>(qn, kn, xf, rinv, mA, mB, (float*)d_out);
}
